// MultiheadAttention_67860483277372
// MI455X (gfx1250) — matrix operand order repaired and checked by reading, not compiled, not run on hardware
//
#include <hip/hip_runtime.h>
#include <hip/hip_bf16.h>
#include <math.h>
#include <stdint.h>

// ---------------------------------------------------------------------------
// MoE grouped GEMM for MI455X (gfx1250, wave32, WMMA + Tensor Data Mover).
//   Pass 0: x (f32) -> bf16 workspace copy (so TDM can stage A tiles)
//   GEMM1:  h = gelu_erf(x_bf16 @ w1[e] + b1[e]) -> bf16 workspace
//   GEMM2:  out = h @ w2[e] + b2[e]              -> fp32 d_out
// A tiles: DMA'd global->LDS by the TDM (tensor_load_to_lds / TENSORcnt),
//          descriptor pads 16B after each 128B row -> bank-spread LDS rows.
// B tiles: f32 weights converted+transposed via packed ds_store_b64.
// Double-buffered LDS, KC=64 so one barrier covers 16 WMMAs per wave.
// ---------------------------------------------------------------------------

typedef __attribute__((ext_vector_type(16))) __bf16 v16bf;
typedef __attribute__((ext_vector_type(8)))  __bf16 v8bf;
typedef __attribute__((ext_vector_type(4)))  __bf16 v4bf;
typedef __attribute__((ext_vector_type(8)))  float  v8f;
typedef __attribute__((ext_vector_type(4)))  unsigned int v4u;
typedef __attribute__((ext_vector_type(4)))  int    v4i;
typedef __attribute__((ext_vector_type(8)))  int    v8i;

#define BM 128
#define BN 128
#define KC 64
#define LDK 72  // 128B data + 16B pad per LDS row; 16B-aligned fragment reads

#if __has_builtin(__builtin_amdgcn_tensor_load_to_lds)
#define USE_TDM 1
#else
#define USE_TDM 0
#endif

__device__ __forceinline__ float gelu_erf(float x) {
    return 0.5f * x * (1.0f + erff(x * 0.70710678118654752440f));
}

#if USE_TDM
// DMA a BM x KC bf16 tile (row stride = strideElems) into LDS at lds_off.
// D# per ISA 8.3/8.4: pad 4 DWORDs after every 32 DWORDs -> 144B LDS rows.
__device__ __forceinline__ void tdm_load_tile(const __bf16* gbase,
                                              unsigned lds_off,
                                              int strideElems) {
    unsigned long long ga = (unsigned long long)(uintptr_t)gbase;
    v4u g0;
    g0[0] = 1u;                                       // count=1 (valid), user mode
    g0[1] = lds_off;                                  // LDS byte address
    g0[2] = (unsigned)(ga & 0xFFFFFFFFu);             // global_addr[31:0]
    g0[3] = (unsigned)((ga >> 32) & 0x01FFFFFFu)      // global_addr[56:32]
          | (2u << 30);                               // type=2 ("image")
    v8i g1;
    g1[0] = (1 << 16)        // data_size = 2 bytes
          | (1 << 20)        // pad_enable
          | (4 << 22)        // pad_interval: 32 DWORDs (128B)
          | (3 << 25);       // pad_amount:   4 DWORDs (16B)
    g1[1] = (KC & 0xFFFF) << 16;                      // tensor_dim0 = KC (tile-exact)
    g1[2] = (BM & 0xFFFF) << 16;                      // tensor_dim1 = BM
    g1[3] = (KC << 16);                               // tile_dim0 = KC
    g1[4] = BM;                                       // tile_dim1 = BM, tile_dim2 = 0
    g1[5] = strideElems;                              // tensor_dim0_stride[31:0]
    g1[6] = 0;
    g1[7] = 0;
    v4i z4 = {0, 0, 0, 0};
#if defined(__clang_major__) && __clang_major__ >= 23
    v8i z8 = {0, 0, 0, 0, 0, 0, 0, 0};
    __builtin_amdgcn_tensor_load_to_lds(g0, g1, z4, z4, z8, 0);
#else
    __builtin_amdgcn_tensor_load_to_lds(g0, g1, z4, z4, 0);
#endif
}
#endif

// f32 -> bf16 bulk convert (8 elements / thread, 16B stores)
__global__ __launch_bounds__(256)
void cvt_f32_bf16(const float* __restrict__ in, __bf16* __restrict__ out,
                  long long n) {
    long long i = ((long long)blockIdx.x * 256 + threadIdx.x) * 8;
    if (i + 8 <= n) {
        float4 a = *(const float4*)(in + i);
        float4 b = *(const float4*)(in + i + 4);
        v8bf v;
        v[0] = (__bf16)a.x; v[1] = (__bf16)a.y; v[2] = (__bf16)a.z; v[3] = (__bf16)a.w;
        v[4] = (__bf16)b.x; v[5] = (__bf16)b.y; v[6] = (__bf16)b.z; v[7] = (__bf16)b.w;
        *(v8bf*)(out + i) = v;
    }
}

template<bool GELU_BF16_OUT>
__global__ __launch_bounds__(256, 2)
void moe_grouped_gemm(const __bf16* __restrict__ A,  // [T][K] bf16
                      const float* __restrict__ W,   // [E][K][N] f32
                      const float* __restrict__ bias,// [E][N]
                      const int* __restrict__ expert_size,
                      void* __restrict__ Out,        // [T][N] bf16 or f32
                      int K, int N, int nE)
{
    __shared__ __attribute__((aligned(16))) __bf16 sA[2][BM][LDK];
    __shared__ __attribute__((aligned(16))) __bf16 sB[2][BN][LDK];

    const int n0   = blockIdx.x * BN;
    const int row0 = blockIdx.y * BM;
    const int t    = threadIdx.x;

    // Resolve this row-tile's expert (group sizes are multiples of BM).
    int e = 0, start = 0;
    for (int i = 0; i < nE; ++i) {
        int s = expert_size[i];
        if (row0 >= start + s) { start += s; e = i + 1; }
    }

    const int lane = t & 31;
    const int wave = t >> 5;
    const int lm   = lane & 15;
    const int lh   = lane >> 4;
    const int wm   = (wave & 3) * 32;
    const int wn   = (wave >> 2) * 64;

    v8f acc[2][4];
#pragma unroll
    for (int i = 0; i < 2; ++i)
#pragma unroll
        for (int j = 0; j < 4; ++j)
            acc[i][j] = (v8f){0.f,0.f,0.f,0.f,0.f,0.f,0.f,0.f};

    const float* We = W + (size_t)e * K * N;
    const int nK = K / KC;

    // ---- tile stagers ----
    auto stageA = [&](int buf, int kbase) {
#if USE_TDM
        if (wave == 0) {
            tdm_load_tile(A + (size_t)row0 * K + kbase,
                          (unsigned)(uintptr_t)&sA[buf][0][0], K);
        }
#else
        const int c8  = (t & 7) * 8;
        const int r0l = t >> 3;
        v8bf v[4];
#pragma unroll
        for (int p = 0; p < 4; ++p)
            v[p] = *(const v8bf*)(A + (size_t)(row0 + r0l + 32 * p) * K + kbase + c8);
#pragma unroll
        for (int p = 0; p < 4; ++p)
            *(v8bf*)&sA[buf][r0l + 32 * p][c8] = v[p];
#endif
    };
    // Each wave owns K-rows [4w..4w+3] and [32+4w..32+4w+3]; lanes cover N.
    // Row-coalesced float4 reads, then 4 K-values packed per column into one
    // ds_store_b64 (transpose f32[K][N] -> bf16 LDS[N][K]).
    auto stageB = [&](int buf, int kbase) {
        const int w4 = (t >> 5) * 4;
        const int cn = (t & 31) * 4;
#pragma unroll
        for (int hh = 0; hh < 2; ++hh) {
            const int kk = hh * 32 + w4;
            float4 v[4];
#pragma unroll
            for (int r = 0; r < 4; ++r)
                v[r] = *(const float4*)(We + (size_t)(kbase + kk + r) * N + n0 + cn);
#pragma unroll
            for (int q = 0; q < 4; ++q) {
                v4bf pk;
                pk[0] = (__bf16)((const float*)&v[0])[q];
                pk[1] = (__bf16)((const float*)&v[1])[q];
                pk[2] = (__bf16)((const float*)&v[2])[q];
                pk[3] = (__bf16)((const float*)&v[3])[q];
                *(v4bf*)&sB[buf][cn + q][kk] = pk;
            }
        }
    };

    // ---- prologue: fill buffer 0 ----
    stageA(0, 0);
    stageB(0, 0);
#if USE_TDM
    if (wave == 0) __builtin_amdgcn_s_wait_tensorcnt(0);
#endif
    __syncthreads();

    for (int kc = 0; kc < nK; ++kc) {
        const int cur = kc & 1;

        // stage next chunk into the other buffer while we compute this one
        if (kc + 1 < nK) {
            stageA(cur ^ 1, (kc + 1) * KC);
            stageB(cur ^ 1, (kc + 1) * KC);
        }

        // ---- two K=32 sub-steps: fragments per ISA layouts + 8 WMMAs each ----
#pragma unroll
        for (int ks = 0; ks < 2; ++ks) {
            const int kb = ks * 32;
            v16bf a[2], b[4];
#pragma unroll
            for (int i = 0; i < 2; ++i) {
                const int r = wm + i * 16 + lm;
                v8bf lo = *(const v8bf*)&sA[cur][r][kb + 8 * lh];
                v8bf hi = *(const v8bf*)&sA[cur][r][kb + 16 + 8 * lh];
#pragma unroll
                for (int q = 0; q < 8; ++q) { a[i][q] = lo[q]; a[i][q + 8] = hi[q]; }
            }
#pragma unroll
            for (int j = 0; j < 4; ++j) {
                const int n = wn + j * 16 + lm;
                v8bf lo = *(const v8bf*)&sB[cur][n][kb + 8 * lh];
                v8bf hi = *(const v8bf*)&sB[cur][n][kb + 16 + 8 * lh];
#pragma unroll
                for (int q = 0; q < 8; ++q) { b[j][q] = lo[q]; b[j][q + 8] = hi[q]; }
            }
#pragma unroll
            for (int j = 0; j < 4; ++j)
#pragma unroll
                for (int i = 0; i < 2; ++i)
                    acc[i][j] = __builtin_amdgcn_wmma_f32_16x16x32_bf16(
                        false, a[i], false, b[j], (short)0, acc[i][j], false, false);
        }

#if USE_TDM
        if (wave == 0) __builtin_amdgcn_s_wait_tensorcnt(0);
#endif
        __syncthreads();
    }

    // ---- epilogue: C layout = VGPR r -> M = r + 8*lh, N = lm ----
#pragma unroll
    for (int j = 0; j < 4; ++j) {
        const int colg = n0 + wn + j * 16 + lm;
        const float bv = bias[(size_t)e * N + colg];
#pragma unroll
        for (int i = 0; i < 2; ++i) {
            const int rowb = row0 + wm + i * 16 + 8 * lh;
#pragma unroll
            for (int r = 0; r < 8; ++r) {
                const float v = acc[i][j][r] + bv;
                if constexpr (GELU_BF16_OUT) {
                    ((__bf16*)Out)[(size_t)(rowb + r) * N + colg] = (__bf16)gelu_erf(v);
                } else {
                    ((float*)Out)[(size_t)(rowb + r) * N + colg] = v;
                }
            }
        }
    }
}

extern "C" void kernel_launch(void* const* d_in, const int* in_sizes, int n_in,
                              void* d_out, int out_size, void* d_ws, size_t ws_size,
                              hipStream_t stream)
{
    const float* x  = (const float*)d_in[0];   // [T][D]
    const int*   es = (const int*)d_in[1];     // [E]
    const float* w1 = (const float*)d_in[2];   // [E][D][H]
    const float* b1 = (const float*)d_in[3];   // [E][H]
    const float* w2 = (const float*)d_in[4];   // [E][H][D]
    const float* b2 = (const float*)d_in[5];   // [E][D]

    const int E = in_sizes[1];
    const int H = in_sizes[3] / E;
    const int D = in_sizes[5] / E;
    const int T = in_sizes[0] / D;

    // ws layout: h_bf16 [T][H]  |  x_bf16 [T][D]
    __bf16* hb = (__bf16*)d_ws;
    __bf16* xb = (__bf16*)((char*)d_ws + (size_t)T * H * sizeof(__bf16));

    dim3 blk(256, 1, 1);

    const long long nx = (long long)T * D;
    cvt_f32_bf16<<<dim3((unsigned)(nx / (256 * 8)), 1, 1), blk, 0, stream>>>(x, xb, nx);

    dim3 g1(H / BN, T / BM, 1);
    moe_grouped_gemm<true ><<<g1, blk, 0, stream>>>(xb, w1, b1, es, hb, D, H, E);

    dim3 g2(D / BN, T / BM, 1);
    moe_grouped_gemm<false><<<g2, blk, 0, stream>>>(hb, w2, b2, es, d_out, H, D, E);
}
